// Sender_70257075027980
// MI455X (gfx1250) — compile-verified, matrix operand order checked by reading, not run on hardware
//
#include <hip/hip_runtime.h>
#include <hip/hip_bf16.h>
#include <math.h>

#define NNODES 100000
#define NEDGES 1600000
#define EPRIME (NEDGES + NNODES)   // with self loops
#define EMB 64
#define HID 128
#define NEG_SLOPE 0.2f

typedef __attribute__((ext_vector_type(16))) _Float16 v16h;
typedef __attribute__((ext_vector_type(8)))  _Float16 v8h;
typedef __attribute__((ext_vector_type(8)))  float    v8f;

// ---------- helpers ----------
__device__ __forceinline__ float atomicMaxF(float* addr, float v) {
    // monotonic bit trick; init must be -inf (0xFF800000)
    if (v >= 0.0f)
        return __int_as_float(atomicMax((int*)addr, __float_as_int(v)));
    else
        return __uint_as_float(atomicMin((unsigned int*)addr, __float_as_uint(v)));
}

__device__ __forceinline__ float lrelu(float v) { return v > 0.0f ? v : NEG_SLOPE * v; }

// ---------- kernels ----------
__global__ void k_zero_scalar(float* p) {
    if (blockIdx.x == 0 && threadIdx.x == 0) p[0] = 0.0f;
}

__global__ void k_reduce_ea(const float* __restrict__ ea, float* __restrict__ out, int n) {
    __shared__ float buf[256];
    float acc = 0.0f;
    for (int i = blockIdx.x * blockDim.x + threadIdx.x; i < n; i += gridDim.x * blockDim.x)
        acc += ea[i];
    buf[threadIdx.x] = acc;
    __syncthreads();
    for (int off = 128; off > 0; off >>= 1) {
        if ((int)threadIdx.x < off) buf[threadIdx.x] += buf[threadIdx.x + off];
        __syncthreads();
    }
    if (threadIdx.x == 0) atomicAdd(out, buf[0]);
}

// layer-1 node transforms: K=2 -> plain FMA, one thread per (node, channel)
__global__ void k_transform1(const float* __restrict__ x,
                             const float* __restrict__ Wl, const float* __restrict__ bl,
                             const float* __restrict__ Wr, const float* __restrict__ br,
                             float* __restrict__ xl, float* __restrict__ xr) {
    int i = blockIdx.x * blockDim.x + threadIdx.x;
    if (i >= NNODES * EMB) return;
    int node = i >> 6, c = i & 63;
    float x0 = x[node * 2], x1 = x[node * 2 + 1];
    xl[i] = fmaf(x0, Wl[c], fmaf(x1, Wl[EMB + c], bl[c]));
    xr[i] = fmaf(x0, Wr[c], fmaf(x1, Wr[EMB + c], br[c]));
}

// init per-layer segment state: m=-inf, denom=0, out=0
__global__ void k_initseg(float* __restrict__ m, float* __restrict__ denom,
                          float* __restrict__ out) {
    int i = blockIdx.x * blockDim.x + threadIdx.x;
    if (i < NNODES) { m[i] = -__builtin_inff(); denom[i] = 0.0f; }
    if (i < NNODES * EMB) out[i] = 0.0f;
}

// edge pass 1: score = att . leaky_relu(xl[src]+xr[dst]+ea*We); segment max via atomics
__global__ void k_score(const int* __restrict__ ei, const float* __restrict__ ea,
                        const float* __restrict__ eaSum,
                        const float* __restrict__ xl, const float* __restrict__ xr,
                        const float* __restrict__ We, const float* __restrict__ att,
                        float* __restrict__ score, float* __restrict__ m) {
    int e = blockIdx.x * blockDim.x + threadIdx.x;
    if (e >= EPRIME) return;
    int s, d; float eav;
    if (e < NEDGES) { s = ei[e]; d = ei[NEDGES + e]; eav = ea[e]; }
    else            { s = d = e - NEDGES; eav = eaSum[0] * (1.0f / (float)NEDGES); }
    const float4* xs = (const float4*)(xl + (size_t)s * EMB);
    const float4* xd = (const float4*)(xr + (size_t)d * EMB);
    const float4* w4 = (const float4*)We;
    const float4* a4 = (const float4*)att;
    float acc = 0.0f;
#pragma unroll
    for (int q = 0; q < EMB / 4; ++q) {
        float4 a = xs[q], b = xd[q], w = w4[q], t = a4[q];
        acc += t.x * lrelu(a.x + b.x + eav * w.x);
        acc += t.y * lrelu(a.y + b.y + eav * w.y);
        acc += t.z * lrelu(a.z + b.z + eav * w.z);
        acc += t.w * lrelu(a.w + b.w + eav * w.w);
    }
    score[e] = acc;
    atomicMaxF(m + d, acc);
}

// edge pass 2: ex = exp(score - m[dst]); segment sum
__global__ void k_exp(const int* __restrict__ ei, float* __restrict__ score,
                      const float* __restrict__ m, float* __restrict__ denom) {
    int e = blockIdx.x * blockDim.x + threadIdx.x;
    if (e >= EPRIME) return;
    int d = (e < NEDGES) ? ei[NEDGES + e] : (e - NEDGES);
    float ex = __expf(score[e] - m[d]);
    score[e] = ex;                 // reuse buffer for ex
    atomicAdd(denom + d, ex);
}

// edge pass 3: out[dst] += (ex/denom[dst]) * xl[src]; 16 threads per edge, float4 each
__global__ void k_agg(const int* __restrict__ ei, const float* __restrict__ ex,
                      const float* __restrict__ denom, const float* __restrict__ xl,
                      float* __restrict__ out) {
    int idx = blockIdx.x * blockDim.x + threadIdx.x;
    if (idx >= EPRIME * 16) return;
    int e = idx >> 4, q = idx & 15;
    int s, d;
    if (e < NEDGES) { s = ei[e]; d = ei[NEDGES + e]; }
    else            { s = d = e - NEDGES; }
    float w = ex[e] / denom[d];
    float4 v = ((const float4*)(xl + (size_t)s * EMB))[q];
    float* o = out + (size_t)d * EMB + q * 4;
    atomicAdd(o + 0, w * v.x);
    atomicAdd(o + 1, w * v.y);
    atomicAdd(o + 2, w * v.z);
    atomicAdd(o + 3, w * v.w);
}

__global__ void k_relu_bias(float* __restrict__ h, const float* __restrict__ b) {
    int i = blockIdx.x * blockDim.x + threadIdx.x;
    if (i >= NNODES * EMB) return;
    h[i] = fmaxf(h[i] + b[i & 63], 0.0f);
}

// layer-2 node transform: Out = H(Nx64) @ W(64x64) + bias via v_wmma_f32_16x16x32_f16.
// Block = 128 (4 waves). W staged once per block into LDS, transposed + f16 (8 KB), so each
// B fragment is two contiguous ds_load_b128. Each A fragment is four global_load_b128.
// Per the ISA A-matrix 16x32 f16 layout, lane halves a[0..7]  = row[kbase+hi*8 + 0..7],
//                                              a[8..15] = row[kbase+hi*8 + 16..23].
__global__ void k_gemm64(const float* __restrict__ H, const float* __restrict__ W,
                         const float* __restrict__ bias, float* __restrict__ Out) {
    __shared__ _Float16 WT[EMB * EMB];    // transposed: WT[col*64 + k]
    for (int i = threadIdx.x; i < EMB * EMB; i += blockDim.x) {
        int k = i >> 6, c = i & 63;
        WT[c * EMB + k] = (_Float16)W[i];
    }
    __syncthreads();

    int wave = threadIdx.x >> 5;
    int lane = threadIdx.x & 31;
    int tile_m = blockIdx.x * 4 + wave;
    if (tile_m >= NNODES / 16) return;    // wave-uniform; EXEC all-1 inside
    int row = lane & 15;
    int hi8 = (lane >> 4) * 8;            // half-wave K offset

    const float* hrow = H + ((size_t)tile_m * 16 + row) * EMB;
    const float4* hp = (const float4*)hrow;

    v16h a0, a1;
    {
        int q0 = hi8 >> 2;                             // chunk0 base /4
        float4 f0 = hp[q0],     f1 = hp[q0 + 1];       // k = hi8 + 0..7
        float4 f2 = hp[q0 + 4], f3 = hp[q0 + 5];       // k = hi8 + 16..23
        a0[0]=(_Float16)f0.x; a0[1]=(_Float16)f0.y; a0[2]=(_Float16)f0.z; a0[3]=(_Float16)f0.w;
        a0[4]=(_Float16)f1.x; a0[5]=(_Float16)f1.y; a0[6]=(_Float16)f1.z; a0[7]=(_Float16)f1.w;
        a0[8]=(_Float16)f2.x; a0[9]=(_Float16)f2.y; a0[10]=(_Float16)f2.z; a0[11]=(_Float16)f2.w;
        a0[12]=(_Float16)f3.x; a0[13]=(_Float16)f3.y; a0[14]=(_Float16)f3.z; a0[15]=(_Float16)f3.w;
        int q1 = (32 + hi8) >> 2;                      // chunk1 base /4
        float4 g0 = hp[q1],     g1 = hp[q1 + 1];
        float4 g2 = hp[q1 + 4], g3 = hp[q1 + 5];
        a1[0]=(_Float16)g0.x; a1[1]=(_Float16)g0.y; a1[2]=(_Float16)g0.z; a1[3]=(_Float16)g0.w;
        a1[4]=(_Float16)g1.x; a1[5]=(_Float16)g1.y; a1[6]=(_Float16)g1.z; a1[7]=(_Float16)g1.w;
        a1[8]=(_Float16)g2.x; a1[9]=(_Float16)g2.y; a1[10]=(_Float16)g2.z; a1[11]=(_Float16)g2.w;
        a1[12]=(_Float16)g3.x; a1[13]=(_Float16)g3.y; a1[14]=(_Float16)g3.z; a1[15]=(_Float16)g3.w;
    }

#pragma unroll
    for (int nt = 0; nt < 4; ++nt) {
        int col = nt * 16 + row;
        const _Float16* wcol = WT + col * EMB;
        // chunk0: k in [hi8+0..7] and [hi8+16..23]; chunk1: +32 — contiguous 16B spans
        v8h b0lo = *(const v8h*)(wcol + hi8);
        v8h b0hi = *(const v8h*)(wcol + hi8 + 16);
        v8h b1lo = *(const v8h*)(wcol + 32 + hi8);
        v8h b1hi = *(const v8h*)(wcol + 32 + hi8 + 16);
        v16h b0, b1;
#pragma unroll
        for (int i = 0; i < 8; ++i) {
            b0[i] = b0lo[i]; b0[8 + i] = b0hi[i];
            b1[i] = b1lo[i]; b1[8 + i] = b1hi[i];
        }
        v8f c = {};
        c = __builtin_amdgcn_wmma_f32_16x16x32_f16(false, a0, false, b0, (short)0, c, false, false);
        c = __builtin_amdgcn_wmma_f32_16x16x32_f16(false, a1, false, b1, (short)0, c, false, false);
        float bv = bias[col];
        // C/D layout: lanes 0-15: VGPR r -> M=r; lanes 16-31: VGPR r -> M=r+8; N=lane%16
#pragma unroll
        for (int r = 0; r < 8; ++r) {
            int mrow = tile_m * 16 + r + hi8;          // hi8 = 8 for upper half-wave
            Out[(size_t)mrow * EMB + col] = c[r] + bv;
        }
    }
}

// head: out[1x128] = h2[target] @ Wf(64x128) + bf
__global__ void k_head(const float* __restrict__ h, const int* __restrict__ tgt,
                       const float* __restrict__ Wf, const float* __restrict__ bf,
                       float* __restrict__ out) {
    int j = threadIdx.x;
    if (j >= HID) return;
    int t = tgt[0];
    float acc = bf[j];
#pragma unroll
    for (int c = 0; c < EMB; ++c)
        acc = fmaf(h[(size_t)t * EMB + c], Wf[(size_t)c * HID + j], acc);
    out[j] = acc;
}

// ---------- launch ----------
extern "C" void kernel_launch(void* const* d_in, const int* in_sizes, int n_in,
                              void* d_out, int out_size, void* d_ws, size_t ws_size,
                              hipStream_t stream) {
    const float* x    = (const float*)d_in[0];
    const int*   ei   = (const int*)d_in[1];
    const float* ea   = (const float*)d_in[2];
    const int*   tgt  = (const int*)d_in[3];
    const float* Wl1  = (const float*)d_in[4];
    const float* bl1  = (const float*)d_in[5];
    const float* Wr1  = (const float*)d_in[6];
    const float* br1  = (const float*)d_in[7];
    const float* We1  = (const float*)d_in[8];
    const float* att1 = (const float*)d_in[9];
    const float* b1   = (const float*)d_in[10];
    const float* Wl2  = (const float*)d_in[11];
    const float* bl2  = (const float*)d_in[12];
    const float* Wr2  = (const float*)d_in[13];
    const float* br2  = (const float*)d_in[14];
    const float* We2  = (const float*)d_in[15];
    const float* att2 = (const float*)d_in[16];
    const float* b2   = (const float*)d_in[17];
    const float* Wf   = (const float*)d_in[18];
    const float* bf   = (const float*)d_in[19];
    float* out = (float*)d_out;

    // workspace layout
    char* ws = (char*)d_ws;
    float* eaSum = (float*)ws;                       // 1 float (padded to 256B)
    float* A   = (float*)(ws + 256);                 // xl   [N*64]
    float* B   = A + (size_t)NNODES * EMB;           // xr   [N*64]
    float* C   = B + (size_t)NNODES * EMB;           // out/h[N*64]
    float* EX  = C + (size_t)NNODES * EMB;           // score/ex [E']
    float* M   = EX + (size_t)EPRIME;                // seg max [N]
    float* DEN = M + (size_t)NNODES;                 // seg sum [N]

    const int T = 256;
    int gNC   = (NNODES * EMB + T - 1) / T;          // N*64 threads
    int gE    = (EPRIME + T - 1) / T;
    int gE16  = (EPRIME * 16 + T - 1) / T;
    int gGemm = ((NNODES / 16) + 3) / 4;             // 4 waves/block, 1 m-tile each

    // edge_attr mean
    k_zero_scalar<<<1, 64, 0, stream>>>(eaSum);
    k_reduce_ea<<<2048, T, 0, stream>>>(ea, eaSum, NEDGES);

    // ---- layer 1 ----
    k_transform1<<<gNC, T, 0, stream>>>(x, Wl1, bl1, Wr1, br1, A, B);
    k_initseg<<<gNC, T, 0, stream>>>(M, DEN, C);
    k_score<<<gE, T, 0, stream>>>(ei, ea, eaSum, A, B, We1, att1, EX, M);
    k_exp<<<gE, T, 0, stream>>>(ei, EX, M, DEN);
    k_agg<<<gE16, T, 0, stream>>>(ei, EX, DEN, A, C);
    k_relu_bias<<<gNC, T, 0, stream>>>(C, b1);       // C = h1

    // ---- layer 2 node transforms via WMMA ----
    k_gemm64<<<gGemm, 128, 0, stream>>>(C, Wl2, bl2, A);   // A = xl2
    k_gemm64<<<gGemm, 128, 0, stream>>>(C, Wr2, br2, B);   // B = xr2

    // ---- layer 2 edge passes ----
    k_initseg<<<gNC, T, 0, stream>>>(M, DEN, C);
    k_score<<<gE, T, 0, stream>>>(ei, ea, eaSum, A, B, We2, att2, EX, M);
    k_exp<<<gE, T, 0, stream>>>(ei, EX, M, DEN);
    k_agg<<<gE16, T, 0, stream>>>(ei, EX, DEN, A, C);
    k_relu_bias<<<gNC, T, 0, stream>>>(C, b2);       // C = h2

    // ---- head ----
    k_head<<<1, HID, 0, stream>>>(C, tgt, Wf, bf, out);
}
